// FNO1DEncoder_74345883894535
// MI455X (gfx1250) — matrix operand order repaired and checked by reading, not compiled, not run on hardware
//
#include <hip/hip_runtime.h>
#include <math.h>

#define NRL    4
#define CH     64
#define MODES_ 32
#define L_     4096
#define LP     4104   // FFT length after PAD=8
#define LPAD   4128   // K-padded stride (multiple of 32), tail zeroed in bases
#define B_     64
#define KB     96     // K panel width staged in LDS (43 * 96 == 4128 exactly)
#define NSTG   43

typedef _Float16 f16;
typedef f16   v16h __attribute__((ext_vector_type(16)));
typedef f16   v8h  __attribute__((ext_vector_type(8)));
typedef float v8f  __attribute__((ext_vector_type(8)));

// ---------------- WMMA fragment helpers (wave32, 16x16x32 f16 -> f32) ----------

// A (16x32, row-major, K contiguous): lane m=(lane&15); 16 contiguous K at
// k=(lane>>4)*16 (matches B fragment order).
__device__ __forceinline__ v16h load_a_frag(const f16* __restrict__ A, int ld,
                                            int m0, int k0, int lane) {
  int lm = lane & 15;
  int lk = (lane >> 4) * 16;
  const f16* p = A + (size_t)(m0 + lm) * ld + k0 + lk;
  v8h lo = *(const v8h*)p;
  v8h hi = *(const v8h*)(p + 8);
  v16h r;
#pragma unroll
  for (int i = 0; i < 8; ++i) { r[i] = lo[i]; r[i + 8] = hi[i]; }
  return r;
}

// B (32x16): element (k,n) stored at Bt[n*ld + k] (K contiguous per column).
// lane n=(lane&15); 16 contiguous K at k=(lane>>4)*16  (same k order as A fragment).
__device__ __forceinline__ v16h load_b_frag(const f16* __restrict__ Bt, int ld,
                                            int n0, int k0, int lane) {
  int ln = lane & 15;
  int kb = (lane >> 4) * 16;
  return *(const v16h*)(Bt + (size_t)(n0 + ln) * ld + k0 + kb);
}

__device__ __forceinline__ v8f wmma_f16(v16h a, v16h b, v8f c) {
  return __builtin_amdgcn_wmma_f32_16x16x32_f16(false, a, false, b,
                                                (short)0, c, false, false);
}

__device__ __forceinline__ v8f vzero8() {
  v8f z = {0.f, 0.f, 0.f, 0.f, 0.f, 0.f, 0.f, 0.f};
  return z;
}

// ---------------- Tensor Data Mover staging (gfx1250 TDM, D# per ISA ch.8) -----
#if defined(__has_builtin)
#if __has_builtin(__builtin_amdgcn_tensor_load_to_lds)
#define HAVE_TDM 1
#endif
#endif

#ifdef HAVE_TDM
typedef unsigned int u32x4 __attribute__((ext_vector_type(4)));
typedef int          i32x8 __attribute__((ext_vector_type(8)));
typedef int          i32x4 __attribute__((ext_vector_type(4)));

// 2D tile load: tensor (dim1=64 rows x dim0=LPAD cols, 2B elems, row stride LPAD),
// tile = 64 x KB starting at gptr, packed contiguously into LDS at lds_off.
__device__ __forceinline__ void tdm_load_panel(const void* gptr, unsigned lds_off) {
  unsigned long long ga = (unsigned long long)(size_t)gptr;
  u32x4 g0;
  g0[0] = 1u;                                           // count=1 (valid user D#)
  g0[1] = lds_off;                                      // lds_addr
  g0[2] = (unsigned)ga;                                 // global_addr[31:0]
  g0[3] = (unsigned)((ga >> 32) & 0x01FFFFFFu) | 0x80000000u; // [56:32] | type=2
  i32x8 g1;
  g1[0] = 0x00010000;                   // workgroup_mask=0, data_size=1 (2 bytes)
  g1[1] = (int)((LPAD & 0xFFFF) << 16); // tensor_dim0 low16 (bits 63:48)
  g1[2] = (int)((CH & 0xFFFF) << 16);   // tensor_dim0 hi=0; tensor_dim1=64 low16
  g1[3] = (int)((KB & 0xFFFF) << 16);   // tensor_dim1 hi=0; tile_dim0=KB
  g1[4] = CH;                           // tile_dim1=64, tile_dim2=0
  g1[5] = LPAD;                         // tensor_dim0_stride[31:0]
  g1[6] = (int)((LPAD & 0xFFFF) << 16); // stride0 hi=0; tensor_dim1_stride low16
  g1[7] = 0;
  i32x4 z4 = {0, 0, 0, 0};
  i32x8 z8 = {0, 0, 0, 0, 0, 0, 0, 0};
  // clang-23 / therock-10.0 signature: (g0, g1, g2, g3, g4, cpol)
  __builtin_amdgcn_tensor_load_to_lds(g0, g1, z4, z4, z8, 0);
}
#endif

// ---------------- basis generation ---------------------------------------------
// fwd  : (64, LPAD)  col n<32: cos(2*pi*m*l/LP), n>=32: -sin(...)   (rfft)
// invT : (LPAD, 64)  k<32: s_m*cos, k>=32: -s_m*sin  (s_0=1/LP, s_m=2/LP; Im(DC)=0)
__global__ void basis_kernel(f16* __restrict__ fwd, f16* __restrict__ invT) {
  int idx = blockIdx.x * blockDim.x + threadIdx.x;
  if (idx >= 64 * LPAD) return;
  int n = idx / LPAD;
  int l = idx - n * LPAD;
  int m = n & 31;
  float fv = 0.f, iv = 0.f;
  if (l < LP) {
    double ang = 6.283185307179586476925286766559 *
                 (double)((m * l) % LP) / (double)LP;
    float c = (float)cos(ang), s = (float)sin(ang);
    float sc = ((m == 0) ? 1.0f : 2.0f) / (float)LP;
    if (n < 32) { fv = c;  iv = sc * c; }
    else        { fv = -s; iv = (m == 0) ? 0.f : -sc * s; }
  }
  fwd[(size_t)n * LPAD + l] = (f16)fv;
  invT[(size_t)l * 64 + n]  = (f16)iv;
}

// ---------------- lift: x(B,1,L) + grid -> activations in both layouts ---------
__global__ void lift_kernel(const float* __restrict__ xin, const float* __restrict__ lw,
                            const float* __restrict__ lb, f16* __restrict__ xa,
                            f16* __restrict__ xaT) {
  int idx = blockIdx.x * blockDim.x + threadIdx.x;
  if (idx >= B_ * CH * LP) return;
  int l = idx % LP;
  int t = idx / LP;
  int o = t & (CH - 1);
  int b = t >> 6;
  float v = 0.f;
  if (l < L_) {
    float g = (float)l * (1.0f / (float)(L_ - 1));
    v = lw[o * 2] * xin[(size_t)b * L_ + l] + lw[o * 2 + 1] * g + lb[o];
  }
  f16 h = (f16)v;
  xa[((size_t)b * CH + o) * LPAD + l] = h;
  xaT[((size_t)b * LPAD + l) * CH + o] = h;
}

// ---------------- per-layer f32 -> f16 weight copy ------------------------------
__global__ void w16_kernel(const float* __restrict__ w, f16* __restrict__ o16) {
  int i = blockIdx.x * blockDim.x + threadIdx.x;
  if (i < CH * CH) o16[i] = (f16)w[i];
}

// ---------------- Stage A: truncated rDFT as GEMM (M=64ch, N=64, K=LPAD) -------
// Xm(b, i, n): n<32 = Re, n>=32 = Im of mode n&31.
// TDM double-buffers 64xKB panels of x[b] and the forward basis in LDS.
__global__ void dft_kernel(const f16* __restrict__ x, const f16* __restrict__ fwd,
                           float* __restrict__ Xm) {
  int b = blockIdx.x;
  int wave = threadIdx.x >> 5, lane = threadIdx.x & 31;
  int mT  = wave >> 1;           // 4 row tiles
  int nT0 = (wave & 1) * 2;      // 2 col tiles per wave
  const f16* xa = x + (size_t)b * CH * LPAD;
  v8f acc0 = vzero8(), acc1 = vzero8();

#ifdef HAVE_TDM
  __shared__ f16 sX[2][CH * KB];
  __shared__ f16 sF[2][CH * KB];
  bool leader = (wave == 0);
  if (leader) {
    tdm_load_panel(xa,  (unsigned)(size_t)&sX[0][0]);
    tdm_load_panel(fwd, (unsigned)(size_t)&sF[0][0]);
    __builtin_amdgcn_s_wait_tensorcnt(0);
  }
  __syncthreads();
  for (int s = 0; s < NSTG; ++s) {
    int buf = s & 1;
    if (leader && (s + 1) < NSTG) {           // prefetch next panel into other buf
      tdm_load_panel(xa  + (size_t)(s + 1) * KB, (unsigned)(size_t)&sX[buf ^ 1][0]);
      tdm_load_panel(fwd + (size_t)(s + 1) * KB, (unsigned)(size_t)&sF[buf ^ 1][0]);
    }
    const f16* lX = &sX[buf][0];
    const f16* lF = &sF[buf][0];
#pragma unroll
    for (int kc = 0; kc < KB / 32; ++kc) {
      int k0 = kc * 32;
      v16h a  = load_a_frag(lX, KB, mT * 16, k0, lane);
      v16h b0 = load_b_frag(lF, KB, nT0 * 16, k0, lane);
      v16h b1 = load_b_frag(lF, KB, (nT0 + 1) * 16, k0, lane);
      acc0 = wmma_f16(a, b0, acc0);
      acc1 = wmma_f16(a, b1, acc1);
    }
    if (leader && (s + 1) < NSTG) __builtin_amdgcn_s_wait_tensorcnt(0);
    __syncthreads();
  }
#else
  for (int k0 = 0; k0 < LPAD; k0 += 32) {
    v16h a  = load_a_frag(xa, LPAD, mT * 16, k0, lane);
    v16h b0 = load_b_frag(fwd, LPAD, nT0 * 16, k0, lane);
    v16h b1 = load_b_frag(fwd, LPAD, (nT0 + 1) * 16, k0, lane);
    acc0 = wmma_f16(a, b0, acc0);
    acc1 = wmma_f16(a, b1, acc1);
  }
#endif

  float* Xb = Xm + (size_t)b * CH * 64;
  int rb = (lane >> 4) * 8, col = lane & 15;
#pragma unroll
  for (int r = 0; r < 8; ++r) {
    Xb[(size_t)(mT * 16 + rb + r) * 64 + nT0 * 16 + col]       = acc0[r];
    Xb[(size_t)(mT * 16 + rb + r) * 64 + (nT0 + 1) * 16 + col] = acc1[r];
  }
}

// ---------------- mode mixing: Om[o,m] = sum_i Xm[i,m] * (wr+i*wi)[i,o,m] ------
__global__ void mix_kernel(const float* __restrict__ Xm, const float* __restrict__ wr,
                           const float* __restrict__ wi, f16* __restrict__ Om) {
  __shared__ float sX[CH * 64];
  int b = blockIdx.x, tid = threadIdx.x;
  const float* Xb = Xm + (size_t)b * CH * 64;
  for (int i = tid; i < CH * 64; i += 256) sX[i] = Xb[i];
  __syncthreads();
  f16* Ob = Om + (size_t)b * CH * 64;
  for (int p = tid; p < CH * MODES_; p += 256) {
    int o = p >> 5, m = p & 31;
    float ar = 0.f, ai = 0.f;
    for (int i = 0; i < CH; ++i) {
      float xr = sX[i * 64 + m];
      float xi = sX[i * 64 + 32 + m];
      size_t wofs = ((size_t)i * CH + o) * MODES_ + m;
      float r = wr[wofs], q = wi[wofs];
      ar += xr * r - xi * q;
      ai += xr * q + xi * r;
    }
    Ob[o * 64 + m]      = (f16)ar;  // pairs with inv rows 0..31  (scaled cos)
    Ob[o * 64 + 32 + m] = (f16)ai;  // pairs with inv rows 32..63 (scaled -sin)
  }
}

// ---------------- Stage C: fused conv1x1 + iDFT + bias + GELU ------------------
// C[o,l] = [conv_w | Om(b)] (64x128)  x  [xT(b) ; invT] (128 x LP)
// Each wave owns two N-tiles (w, w+8) so A-fragments are reused 2x.
__global__ void layer_kernel(const f16* __restrict__ cw16, const f16* __restrict__ Om,
                             const f16* __restrict__ xT, const f16* __restrict__ invT,
                             const float* __restrict__ cb, f16* __restrict__ xn,
                             f16* __restrict__ xnT, float* __restrict__ out, int flags) {
  int b = blockIdx.x;
  int wave = threadIdx.x >> 5, lane = threadIdx.x & 31;
  int nTa = blockIdx.y * 16 + wave;
  int nTb = nTa + 8;
  if (nTa * 16 >= LP) return;                 // 257 tiles cover LP=4104
  bool vb = (nTb * 16) < LP;                  // wave-uniform
  int n0a = nTa * 16, n0b = nTb * 16;
  const f16* xTb = xT + (size_t)b * LPAD * CH;
  const f16* Ob  = Om + (size_t)b * CH * 64;
  v8f acc[2][4];
#pragma unroll
  for (int t = 0; t < 2; ++t)
#pragma unroll
    for (int mT = 0; mT < 4; ++mT) acc[t][mT] = vzero8();

#pragma unroll
  for (int kc = 0; kc < 4; ++kc) {
    int k0 = kc * 32;
    v16h bA = (kc < 2) ? load_b_frag(xTb, CH, n0a, k0, lane)
                       : load_b_frag(invT, 64, n0a, k0 - 64, lane);
    v16h bB = bA;
    if (vb) bB = (kc < 2) ? load_b_frag(xTb, CH, n0b, k0, lane)
                          : load_b_frag(invT, 64, n0b, k0 - 64, lane);
    const f16* A = (kc < 2) ? cw16 : Ob;
    int ak = (kc & 1) * 32;
#pragma unroll
    for (int mT = 0; mT < 4; ++mT) {
      v16h af = load_a_frag(A, 64, mT * 16, ak, lane);
      acc[0][mT] = wmma_f16(af, bA, acc[0][mT]);
      if (vb) acc[1][mT] = wmma_f16(af, bB, acc[1][mT]);
    }
  }

  int col = lane & 15, rb = (lane >> 4) * 8;
  bool hasAct = (flags & 1) != 0, isLast = (flags & 2) != 0;
#pragma unroll
  for (int t = 0; t < 2; ++t) {
    if (t == 1 && !vb) break;
    int l = (t == 0 ? n0a : n0b) + col;
    if (l >= LP) continue;
#pragma unroll
    for (int mT = 0; mT < 4; ++mT) {
#pragma unroll
      for (int r = 0; r < 8; ++r) {
        int o = mT * 16 + rb + r;
        float v = acc[t][mT][r] + cb[o];
        if (hasAct) v = 0.5f * v * (1.0f + erff(v * 0.70710678118654752f));
        if (isLast) {
          if (l < L_) out[((size_t)b * CH + o) * L_ + l] = v;
        } else {
          f16 h = (f16)v;
          xn[((size_t)b * CH + o) * LPAD + l] = h;
          xnT[((size_t)b * LPAD + l) * CH + o] = h;
        }
      }
    }
  }
}

// ---------------- host side -----------------------------------------------------
extern "C" void kernel_launch(void* const* d_in, const int* in_sizes, int n_in,
                              void* d_out, int out_size, void* d_ws, size_t ws_size,
                              hipStream_t stream) {
  const float* x   = (const float*)d_in[0];
  const float* lw  = (const float*)d_in[1];
  const float* lb  = (const float*)d_in[2];
  const float* cw  = (const float*)d_in[3];
  const float* cb  = (const float*)d_in[4];
  const float* swr = (const float*)d_in[5];
  const float* swi = (const float*)d_in[6];
  float* out = (float*)d_out;

  char* ws = (char*)d_ws;
  size_t off = 0;
  auto alloc = [&](size_t bytes) -> void* {
    void* p = ws + off;
    off = (off + bytes + 255) & ~(size_t)255;
    return p;
  };
  f16*   fwd  = (f16*)alloc((size_t)64 * LPAD * sizeof(f16));
  f16*   invT = (f16*)alloc((size_t)LPAD * 64 * sizeof(f16));
  f16*   cw16 = (f16*)alloc((size_t)CH * CH * sizeof(f16));
  f16*   Om   = (f16*)alloc((size_t)B_ * CH * 64 * sizeof(f16));
  float* Xm   = (float*)alloc((size_t)B_ * CH * 64 * sizeof(float));
  f16*   xA   = (f16*)alloc((size_t)B_ * CH * LPAD * sizeof(f16));
  f16*   xAT  = (f16*)alloc((size_t)B_ * LPAD * CH * sizeof(f16));
  f16*   xB   = (f16*)alloc((size_t)B_ * CH * LPAD * sizeof(f16));
  f16*   xBT  = (f16*)alloc((size_t)B_ * LPAD * CH * sizeof(f16));

  basis_kernel<<<(64 * LPAD + 255) / 256, 256, 0, stream>>>(fwd, invT);

  {
    int total = B_ * CH * LP;
    lift_kernel<<<(total + 255) / 256, 256, 0, stream>>>(x, lw, lb, xA, xAT);
  }

  f16 *cur = xA, *curT = xAT, *nxt = xB, *nxtT = xBT;
  for (int k = 0; k < NRL; ++k) {
    w16_kernel<<<(CH * CH + 255) / 256, 256, 0, stream>>>(cw + (size_t)k * CH * CH, cw16);
    dft_kernel<<<B_, 256, 0, stream>>>(cur, fwd, Xm);
    mix_kernel<<<B_, 256, 0, stream>>>(Xm, swr + (size_t)k * CH * CH * MODES_,
                                       swi + (size_t)k * CH * CH * MODES_, Om);
    int flags = ((k < NRL - 1) ? 1 : 0) | ((k == NRL - 1) ? 2 : 0);
    layer_kernel<<<dim3(B_, 17), 256, 0, stream>>>(cw16, Om, curT, invT,
                                                   cb + (size_t)k * CH, nxt, nxtT,
                                                   out, flags);
    f16* t;
    t = cur;  cur  = nxt;  nxt  = t;
    t = curT; curT = nxtT; nxtT = t;
  }
  (void)in_sizes; (void)n_in; (void)out_size; (void)ws_size;
}
